// Codebook_38886633898343
// MI455X (gfx1250) — matrix operand order repaired and checked by reading, not compiled, not run on hardware
//
#include <hip/hip_runtime.h>

// ---------------------------------------------------------------------------
// Problem constants (from the reference)
// ---------------------------------------------------------------------------
#define BQ      512      // batch
#define NBOOK   65536    // codebook entries
#define DDIM    256      // feature dim (both rot and trans)
#define NPART   64       // column partials per row (gridDim.y)
#define A_STR   264      // padded LDS row stride (bf16 elements)
#define MTILE   32       // rows per block (two 16-row WMMA tiles share each B tile)

typedef __attribute__((ext_vector_type(16))) __bf16 v16bf;
typedef __attribute__((ext_vector_type(8)))  float  v8f;

struct U32x8 { uint4 lo, hi; };   // 32 bytes == sizeof(v16bf)

// ---- bf16 helpers (pure bit manipulation, round-to-nearest-even) ----------
__device__ __forceinline__ unsigned short f2bf_bits(float f) {
    unsigned u = __builtin_bit_cast(unsigned, f);
    unsigned r = u + 0x7FFFu + ((u >> 16) & 1u);
    return (unsigned short)(r >> 16);
}
__device__ __forceinline__ float bfbits2f(unsigned short s) {
    unsigned u = ((unsigned)s) << 16;
    return __builtin_bit_cast(float, u);
}

__device__ __forceinline__ v16bf lds_v16bf(const unsigned short* p) {
    U32x8 u;
    u.lo = *(const uint4*)(p);
    u.hi = *(const uint4*)(p + 8);
    return __builtin_bit_cast(v16bf, u);
}

// ---------------------------------------------------------------------------
// Kernel 0: per-row L2-normalize the rotation queries (512 rows x 256)
// ---------------------------------------------------------------------------
__global__ __launch_bounds__(256) void normalize_kernel(
    const float* __restrict__ q, float* __restrict__ outN)
{
    const int row = blockIdx.x;
    const int t   = threadIdx.x;
    float x = q[row * DDIM + t];
    float s = x * x;
    #pragma unroll
    for (int m = 16; m >= 1; m >>= 1) s += __shfl_xor(s, m, 32);
    __shared__ float red[8];
    if ((t & 31) == 0) red[t >> 5] = s;
    __syncthreads();
    float tot = red[0] + red[1] + red[2] + red[3] + red[4] + red[5] + red[6] + red[7];
    outN[row * DDIM + t] = x / sqrtf(tot);
}

// ---------------------------------------------------------------------------
// Kernel 1: one-time codebook preprocessing.
//   f32 [N,256] -> split bf16 hi/lo arrays [N,256] + per-row ||b||^2.
//   grid = (N/16, 2 books), block = 256 (8 waves, each wave does 2 rows)
// ---------------------------------------------------------------------------
__global__ __launch_bounds__(256) void convert_book_kernel(
    const float* __restrict__ rotBookF, const float* __restrict__ trnBookF,
    unsigned short* __restrict__ rotHi, unsigned short* __restrict__ rotLo,
    unsigned short* __restrict__ trnHi, unsigned short* __restrict__ trnLo,
    float* __restrict__ b2rot, float* __restrict__ b2trn)
{
    const int book = blockIdx.y;
    const float* __restrict__ src = book ? trnBookF : rotBookF;
    unsigned short* __restrict__ dh = book ? trnHi : rotHi;
    unsigned short* __restrict__ dl = book ? trnLo : rotLo;
    float* __restrict__ b2 = book ? b2trn : b2rot;

    const int wave = threadIdx.x >> 5;
    const int lane = threadIdx.x & 31;

    #pragma unroll
    for (int rr = 0; rr < 2; ++rr) {
        const int row = blockIdx.x * 16 + wave * 2 + rr;
        const size_t rb = (size_t)row * DDIM + lane * 8;
        const float4* p = (const float4*)(src + rb);
        float4 a = p[0], b = p[1];
        float xs[8] = {a.x, a.y, a.z, a.w, b.x, b.y, b.z, b.w};
        uint4 hi, lo;
        unsigned hw[4], lw[4];
        float s = 0.0f;
        #pragma unroll
        for (int k = 0; k < 4; ++k) {
            float x0 = xs[2 * k], x1 = xs[2 * k + 1];
            s += x0 * x0 + x1 * x1;
            unsigned short h0 = f2bf_bits(x0), h1 = f2bf_bits(x1);
            unsigned short l0 = f2bf_bits(x0 - bfbits2f(h0));
            unsigned short l1 = f2bf_bits(x1 - bfbits2f(h1));
            hw[k] = (unsigned)h0 | ((unsigned)h1 << 16);
            lw[k] = (unsigned)l0 | ((unsigned)l1 << 16);
        }
        hi.x = hw[0]; hi.y = hw[1]; hi.z = hw[2]; hi.w = hw[3];
        lo.x = lw[0]; lo.y = lw[1]; lo.z = lw[2]; lo.w = lw[3];
        *(uint4*)(dh + rb) = hi;
        *(uint4*)(dl + rb) = lo;
        #pragma unroll
        for (int m = 16; m >= 1; m >>= 1) s += __shfl_xor(s, m, 32);
        if (lane == 0) b2[row] = s;
    }
}

// ---------------------------------------------------------------------------
// Kernel 2: fused dual GEMM (split-bf16 WMMA) + per-tile argmax.
//   M=32 per block: each loaded B tile feeds two 16-row A tiles.
//   K loop unrolled 2x only: bounds live v16bf tiles so the allocation stays
//   under the 512-VGPR/wave budget (round 4's full unroll spilled to scratch).
//   grid = (16 row-blocks of 32, 64 column groups of 1024), block = 256 (8 waves)
// ---------------------------------------------------------------------------
__global__ __launch_bounds__(256) void score_argmax_kernel(
    const float* __restrict__ qrotN,           // [512,256] normalized rot queries
    const float* __restrict__ qtrans,          // [512,256]
    const unsigned short* __restrict__ rotHi,  // [65536,256] bf16 bits
    const unsigned short* __restrict__ rotLo,
    const unsigned short* __restrict__ trnHi,
    const unsigned short* __restrict__ trnLo,
    const float* __restrict__ b2trn,           // [65536] ||b||^2
    float* __restrict__ rotVal, int* __restrict__ rotIdx,   // [512,64]
    float* __restrict__ trnVal, int* __restrict__ trnIdx)   // [512,64]
{
    // Query tiles staged as pre-split bf16 hi/lo (conversion once per block)
    __shared__ __align__(16) unsigned short ldsRhi[MTILE * A_STR];
    __shared__ __align__(16) unsigned short ldsRlo[MTILE * A_STR];
    __shared__ __align__(16) unsigned short ldsThi[MTILE * A_STR];
    __shared__ __align__(16) unsigned short ldsTlo[MTILE * A_STR];
    __shared__ float redV[2][8][MTILE];
    __shared__ int   redI[2][8][MTILE];

    const int tid = threadIdx.x;
    const int m0  = blockIdx.x * MTILE;

    // ---- stage 32x256 query tiles (rot + trans), split into hi/lo bf16 ----
    #pragma unroll
    for (int i = 0; i < MTILE; ++i) {
        int row = i, col = tid;
        float xr = qrotN [(m0 + row) * DDIM + col];
        float xt = qtrans[(m0 + row) * DDIM + col];
        unsigned short hr = f2bf_bits(xr);
        unsigned short ht = f2bf_bits(xt);
        ldsRhi[row * A_STR + col] = hr;
        ldsRlo[row * A_STR + col] = f2bf_bits(xr - bfbits2f(hr));
        ldsThi[row * A_STR + col] = ht;
        ldsTlo[row * A_STR + col] = f2bf_bits(xt - bfbits2f(ht));
    }
    __syncthreads();

    const int wave = tid >> 5;
    const int lane = tid & 31;
    const int half = lane >> 4;      // selects K sub-segment per WMMA layout
    const int lnm  = lane & 15;      // A: row M;  B: column N

    float bvR1[8], bvT1[8], bvR2[8], bvT2[8];
    int   biR1[8], biT1[8], biR2[8], biT2[8];
    #pragma unroll
    for (int r = 0; r < 8; ++r) {
        bvR1[r] = -3.4e38f; bvT1[r] = -3.4e38f; biR1[r] = 0; biT1[r] = 0;
        bvR2[r] = -3.4e38f; bvT2[r] = -3.4e38f; biR2[r] = 0; biT2[r] = 0;
    }

    for (int sub = 0; sub < 8; ++sub) {
        // column (codebook row) owned by this lane for this subtile
        const int n = (blockIdx.y * 8 + sub) * 128 + wave * 16 + lnm;
        const unsigned short* __restrict__ rh = rotHi + (size_t)n * DDIM;
        const unsigned short* __restrict__ rl = rotLo + (size_t)n * DDIM;
        const unsigned short* __restrict__ th = trnHi + (size_t)n * DDIM;
        const unsigned short* __restrict__ tl = trnLo + (size_t)n * DDIM;
        if (n + 128 < NBOOK) {   // warm next subtile's rows
            __builtin_prefetch(rotHi + (size_t)(n + 128) * DDIM, 0, 1);
            __builtin_prefetch(trnHi + (size_t)(n + 128) * DDIM, 0, 1);
        }

        v8f cR1 = {}, cR2 = {};
        v8f cT1 = {}, cT2 = {};

        #pragma unroll 2
        for (int ck = 0; ck < 8; ++ck) {
            const int base  = ck * 32;
            const int aoff1 = lnm * A_STR + base + 16 * half;          // tile1: rows 0..15
            const int aoff2 = (lnm + 16) * A_STR + base + 16 * half;   // tile2: rows 16..31
            const int boff  = base + 16 * half;                       // B: one 16-elem segment

            U32x8 u;

            // ---------------- rot ----------------
            u.lo = *(const uint4*)(rh + boff);
            u.hi = *(const uint4*)(rh + boff + 8);
            v16bf bh = __builtin_bit_cast(v16bf, u);
            u.lo = *(const uint4*)(rl + boff);
            u.hi = *(const uint4*)(rl + boff + 8);
            v16bf bl = __builtin_bit_cast(v16bf, u);

            v16bf a1h = lds_v16bf(ldsRhi + aoff1);
            v16bf a1l = lds_v16bf(ldsRlo + aoff1);
            v16bf a2h = lds_v16bf(ldsRhi + aoff2);
            v16bf a2l = lds_v16bf(ldsRlo + aoff2);

            cR1 = __builtin_amdgcn_wmma_f32_16x16x32_bf16(false, a1h, false, bh, (short)0, cR1, false, false);
            cR1 = __builtin_amdgcn_wmma_f32_16x16x32_bf16(false, a1h, false, bl, (short)0, cR1, false, false);
            cR1 = __builtin_amdgcn_wmma_f32_16x16x32_bf16(false, a1l, false, bh, (short)0, cR1, false, false);
            cR2 = __builtin_amdgcn_wmma_f32_16x16x32_bf16(false, a2h, false, bh, (short)0, cR2, false, false);
            cR2 = __builtin_amdgcn_wmma_f32_16x16x32_bf16(false, a2h, false, bl, (short)0, cR2, false, false);
            cR2 = __builtin_amdgcn_wmma_f32_16x16x32_bf16(false, a2l, false, bh, (short)0, cR2, false, false);

            // ---------------- trans ----------------
            u.lo = *(const uint4*)(th + boff);
            u.hi = *(const uint4*)(th + boff + 8);
            bh = __builtin_bit_cast(v16bf, u);
            u.lo = *(const uint4*)(tl + boff);
            u.hi = *(const uint4*)(tl + boff + 8);
            bl = __builtin_bit_cast(v16bf, u);

            a1h = lds_v16bf(ldsThi + aoff1);
            a1l = lds_v16bf(ldsTlo + aoff1);
            a2h = lds_v16bf(ldsThi + aoff2);
            a2l = lds_v16bf(ldsTlo + aoff2);

            cT1 = __builtin_amdgcn_wmma_f32_16x16x32_bf16(false, a1h, false, bh, (short)0, cT1, false, false);
            cT1 = __builtin_amdgcn_wmma_f32_16x16x32_bf16(false, a1h, false, bl, (short)0, cT1, false, false);
            cT1 = __builtin_amdgcn_wmma_f32_16x16x32_bf16(false, a1l, false, bh, (short)0, cT1, false, false);
            cT2 = __builtin_amdgcn_wmma_f32_16x16x32_bf16(false, a2h, false, bh, (short)0, cT2, false, false);
            cT2 = __builtin_amdgcn_wmma_f32_16x16x32_bf16(false, a2h, false, bl, (short)0, cT2, false, false);
            cT2 = __builtin_amdgcn_wmma_f32_16x16x32_bf16(false, a2l, false, bh, (short)0, cT2, false, false);
        }

        const float b2 = b2trn[n];

        // merge this subtile's scores into per-lane running argmax
        // (C layout: c[r] = row m = r + 8*half (+16 for tile2), col n)
        #pragma unroll
        for (int r = 0; r < 8; ++r) {
            float v;
            v = cR1[r];
            if (v > bvR1[r] || (v == bvR1[r] && n < biR1[r])) { bvR1[r] = v; biR1[r] = n; }
            v = cR2[r];
            if (v > bvR2[r] || (v == bvR2[r] && n < biR2[r])) { bvR2[r] = v; biR2[r] = n; }
            v = 2.0f * cT1[r] - b2;
            if (v > bvT1[r] || (v == bvT1[r] && n < biT1[r])) { bvT1[r] = v; biT1[r] = n; }
            v = 2.0f * cT2[r] - b2;
            if (v > bvT2[r] || (v == bvT2[r] && n < biT2[r])) { bvT2[r] = v; biT2[r] = n; }
        }
    }

    // ---- reduce over the 16 columns (lanes within each half), ties -> low idx
    #pragma unroll
    for (int r = 0; r < 8; ++r) {
        #pragma unroll
        for (int m = 1; m <= 8; m <<= 1) {
            float ov; int oi;
            ov = __shfl_xor(bvR1[r], m, 32); oi = __shfl_xor(biR1[r], m, 32);
            if (ov > bvR1[r] || (ov == bvR1[r] && oi < biR1[r])) { bvR1[r] = ov; biR1[r] = oi; }
            ov = __shfl_xor(bvR2[r], m, 32); oi = __shfl_xor(biR2[r], m, 32);
            if (ov > bvR2[r] || (ov == bvR2[r] && oi < biR2[r])) { bvR2[r] = ov; biR2[r] = oi; }
            ov = __shfl_xor(bvT1[r], m, 32); oi = __shfl_xor(biT1[r], m, 32);
            if (ov > bvT1[r] || (ov == bvT1[r] && oi < biT1[r])) { bvT1[r] = ov; biT1[r] = oi; }
            ov = __shfl_xor(bvT2[r], m, 32); oi = __shfl_xor(biT2[r], m, 32);
            if (ov > bvT2[r] || (ov == bvT2[r] && oi < biT2[r])) { bvT2[r] = ov; biT2[r] = oi; }
        }
    }

    // ---- cross-wave reduction through LDS: one partial per row per block ----
    if (lnm == 0) {
        #pragma unroll
        for (int r = 0; r < 8; ++r) {
            int row1 = 8 * half + r;
            int row2 = 16 + 8 * half + r;
            redV[0][wave][row1] = bvR1[r]; redI[0][wave][row1] = biR1[r];
            redV[0][wave][row2] = bvR2[r]; redI[0][wave][row2] = biR2[r];
            redV[1][wave][row1] = bvT1[r]; redI[1][wave][row1] = biT1[r];
            redV[1][wave][row2] = bvT2[r]; redI[1][wave][row2] = biT2[r];
        }
    }
    __syncthreads();
    if (tid < 64) {
        int mat = tid >> 5, row = tid & 31;
        float bv = redV[mat][0][row]; int bi = redI[mat][0][row];
        #pragma unroll
        for (int w = 1; w < 8; ++w) {
            float ov = redV[mat][w][row]; int oi = redI[mat][w][row];
            if (ov > bv || (ov == bv && oi < bi)) { bv = ov; bi = oi; }
        }
        int grow = m0 + row;
        if (mat == 0) { rotVal[grow * NPART + blockIdx.y] = bv; rotIdx[grow * NPART + blockIdx.y] = bi; }
        else          { trnVal[grow * NPART + blockIdx.y] = bv; trnIdx[grow * NPART + blockIdx.y] = bi; }
    }
}

// ---------------------------------------------------------------------------
// Kernel 3: final reduction over the 64 partials + label gather + output
//   grid = 512 (rows), block = 64 (wave0 -> rot, wave1 -> trans)
// ---------------------------------------------------------------------------
__global__ __launch_bounds__(64) void reduce_write_kernel(
    const float* __restrict__ rotVal, const int* __restrict__ rotIdx,
    const float* __restrict__ trnVal, const int* __restrict__ trnIdx,
    const float* __restrict__ rot_book,    // [N,1]
    const float* __restrict__ trans_book,  // [N,3]
    float* __restrict__ out)               // [512 rot | 512*3 trans]
{
    const int row  = blockIdx.x;
    const int wave = threadIdx.x >> 5;
    const int lane = threadIdx.x & 31;
    const float* V = wave ? trnVal : rotVal;
    const int*   I = wave ? trnIdx : rotIdx;

    float v0 = V[row * NPART + lane];       int i0 = I[row * NPART + lane];
    float v1 = V[row * NPART + lane + 32];  int i1 = I[row * NPART + lane + 32];
    if (v1 > v0 || (v1 == v0 && i1 < i0)) { v0 = v1; i0 = i1; }
    #pragma unroll
    for (int m = 16; m >= 1; m >>= 1) {
        float ov = __shfl_xor(v0, m, 32); int oi = __shfl_xor(i0, m, 32);
        if (ov > v0 || (ov == v0 && oi < i0)) { v0 = ov; i0 = oi; }
    }
    if (lane == 0) {
        if (wave == 0) {
            out[row] = rot_book[i0];                        // rotations [B,1,1]
        } else {
            out[BQ + 3 * row + 0] = trans_book[3 * i0 + 0]; // translations [B,3]
            out[BQ + 3 * row + 1] = trans_book[3 * i0 + 1];
            out[BQ + 3 * row + 2] = trans_book[3 * i0 + 2];
        }
    }
}

// ---------------------------------------------------------------------------
// Host launcher
// ---------------------------------------------------------------------------
extern "C" void kernel_launch(void* const* d_in, const int* in_sizes, int n_in,
                              void* d_out, int out_size, void* d_ws, size_t ws_size,
                              hipStream_t stream) {
    const float* z_rot_q     = (const float*)d_in[0];
    const float* z_trans_q   = (const float*)d_in[1];
    const float* z_rot_book  = (const float*)d_in[2];
    const float* z_trans_book= (const float*)d_in[3];
    const float* rot_book    = (const float*)d_in[4];
    const float* trans_book  = (const float*)d_in[5];
    // d_in[6] = k (always 1 in this problem)

    float* ws     = (float*)d_ws;
    float* qnorm  = ws;                          // 512*256
    float* b2rot  = qnorm + BQ * DDIM;           // 65536 (unused scratch)
    float* b2trn  = b2rot + NBOOK;               // 65536
    float* rotVal = b2trn + NBOOK;               // 512*64
    int*   rotIdx = (int*)(rotVal + BQ * NPART);
    float* trnVal = (float*)(rotIdx + BQ * NPART);
    int*   trnIdx = (int*)(trnVal + BQ * NPART);
    unsigned short* rotHi = (unsigned short*)(trnIdx + BQ * NPART);
    unsigned short* rotLo = rotHi + (size_t)NBOOK * DDIM;
    unsigned short* trnHi = rotLo + (size_t)NBOOK * DDIM;
    unsigned short* trnLo = trnHi + (size_t)NBOOK * DDIM;
    float* out    = (float*)d_out;

    normalize_kernel<<<BQ, 256, 0, stream>>>(z_rot_q, qnorm);

    dim3 cgrid(NBOOK / 16, 2);
    convert_book_kernel<<<cgrid, 256, 0, stream>>>(
        z_rot_book, z_trans_book, rotHi, rotLo, trnHi, trnLo, b2rot, b2trn);

    dim3 grid(BQ / MTILE, NPART);   // 16 x 64
    score_argmax_kernel<<<grid, 256, 0, stream>>>(
        qnorm, z_trans_q, rotHi, rotLo, trnHi, trnLo, b2trn,
        rotVal, rotIdx, trnVal, trnIdx);

    reduce_write_kernel<<<BQ, 64, 0, stream>>>(
        rotVal, rotIdx, trnVal, trnIdx, rot_book, trans_book, out);
}
